// MyAttentionProcessor_80212809220328
// MI455X (gfx1250) — matrix operand order repaired and checked by reading, not compiled, not run on hardware
//
#include <hip/hip_runtime.h>
#include <hip/hip_bf16.h>

// ---------------------------------------------------------------------------
// CDNA5 (gfx1250) fused cross-attention, bf16 WMMA pipeline, v4.
//   B=16, S=4096, SKV=77 (padded 96), HID=1280, CDIM=768, HEADS=8, d=160
// v4: O-projection fused into the attention kernel via a 40KB LDS attn tile
//     (92KB static LDS / block, CDNA5 allows up to 320KB per workgroup).
//     Eliminates the 336MB attn-intermediate HBM round trip.
// ---------------------------------------------------------------------------

typedef __attribute__((ext_vector_type(16))) __bf16 v16bf;
typedef __attribute__((ext_vector_type(8)))  float  v8f;
typedef __attribute__((ext_vector_type(4)))  float  v4f;
typedef __attribute__((ext_vector_type(4)))  __bf16 v4bf;

#define NB     16
#define SEQ    4096
#define SKV    77
#define SKV_P  96
#define HID    1280
#define CDIM   768
#define HEADS  8
#define DHEAD  160

// native RNE f32 -> bf16 (v_cvt_pk_bf16_f32 class on gfx1250)
__device__ __forceinline__ __bf16 to_bf16(float f) { return (__bf16)f; }

__device__ __forceinline__ v8f wmma_bf16(v16bf a, v16bf b, v8f c) {
    // (neg_a, A, neg_b, B, c_mod, C, reuse_a, reuse_b)
    return __builtin_amdgcn_wmma_f32_16x16x32_bf16(false, a, false, b, (short)0, c,
                                                   false, false);
}

// A-operand 16x32 (MxK) from row-major bf16 (ISA 16-bit A layout:
// lanes 0-15 -> K {0..7,16..23}, lanes 16-31 -> K {8..15,24..31})
__device__ __forceinline__ v16bf load_a_bf16(const __bf16* base, int ld, int lane, int k0) {
    int m = lane & 15;
    int koff = (lane >> 4) * 8;
    const __bf16* p = base + (size_t)m * ld + k0 + koff;
    v16bf a;
#pragma unroll
    for (int e = 0; e < 8; ++e) a[e] = p[e];
#pragma unroll
    for (int e = 0; e < 8; ++e) a[8 + e] = p[16 + e];
    return a;
}

// B-operand 32x16 (KxN), B[k,n] = W[n,k] i.e. C = A @ W^T.
// Lane (n = lane&15) pulls K from row n of W
// (lanes 0-15 -> K {0..7,16..23}, lanes 16-31 -> K {8..15,24..31}).
__device__ __forceinline__ v16bf load_b_bf16(const __bf16* wbase, int ld, int lane, int k0) {
    int n = lane & 15;
    int koff = (lane >> 4) * 8;
    const __bf16* p = wbase + (size_t)n * ld + k0 + koff;
    v16bf b;
#pragma unroll
    for (int e = 0; e < 8; ++e) b[e] = p[e];
#pragma unroll
    for (int e = 0; e < 8; ++e) b[8 + e] = p[16 + e];
    return b;
}

// ---------------------------------------------------------------------------
// K0a: vectorized fp32 -> bf16 conversion (x4), n must be a multiple of 4
// ---------------------------------------------------------------------------
__global__ void cvt_bf16x4_kernel(const float* __restrict__ src, __bf16* __restrict__ dst,
                                  size_t n4) {
    const v4f* s4 = (const v4f*)src;
    v4bf* d4 = (v4bf*)dst;
    size_t i = (size_t)blockIdx.x * blockDim.x + threadIdx.x;
    size_t stride = (size_t)gridDim.x * blockDim.x;
    for (; i < n4; i += stride) {
        v4f x = s4[i];
        v4bf y;
#pragma unroll
        for (int e = 0; e < 4; ++e) y[e] = to_bf16(x[e]);
        d4[i] = y;
    }
}

// ---------------------------------------------------------------------------
// K0b: encoder states -> bf16, zero-padded from 77 to 96 tokens
//      dst layout [NB][SKV_P][CDIM]
// ---------------------------------------------------------------------------
__global__ void cvt_enc_kernel(const float* __restrict__ enc, __bf16* __restrict__ dst) {
    size_t n4 = (size_t)NB * SKV_P * CDIM / 4;
    v4bf* d4 = (v4bf*)dst;
    size_t i = (size_t)blockIdx.x * blockDim.x + threadIdx.x;
    size_t stride = (size_t)gridDim.x * blockDim.x;
    for (; i < n4; i += stride) {
        size_t elem = i * 4;
        size_t c = elem % CDIM;
        size_t j = (elem / CDIM) % SKV_P;
        size_t b = elem / ((size_t)CDIM * SKV_P);
        v4bf y;
        if (j < SKV) {
            const float* p = enc + (b * SKV + j) * CDIM + c;
#pragma unroll
            for (int e = 0; e < 4; ++e) y[e] = to_bf16(p[e]);
        } else {
#pragma unroll
            for (int e = 0; e < 4; ++e) y[e] = (__bf16)0.0f;
        }
        d4[i] = y;
    }
}

// ---------------------------------------------------------------------------
// K1: K/V projection. One wave = 16 tokens x 64 output channels, 8 wmma
// accumulators (4 n-tiles x {K,V}); all 8 B operands issued before the
// wmma chain so waits retire at partial loadcnt.
// k stored row-major [B, SKV_P, HID]; v stored transposed [B, HID, SKV_P].
// ---------------------------------------------------------------------------
__global__ __launch_bounds__(128)
void kv_proj_kernel(const __bf16* __restrict__ encb, const __bf16* __restrict__ wk,
                    const __bf16* __restrict__ wv, __bf16* __restrict__ kout,
                    __bf16* __restrict__ vtout) {
    int lane = threadIdx.x & 31;
    int wave = threadIdx.x >> 5;
    int task = blockIdx.x * 4 + wave;            // [NB][6 jt][20 ntg]
    int ntg = task % 20;
    int t2  = task / 20;
    int jt  = t2 % 6;
    int b   = t2 / 6;
    if (b >= NB) return;

    const __bf16* abase = encb + ((size_t)b * SKV_P + jt * 16) * CDIM;
    const __bf16* wkb = wk + (size_t)ntg * 64 * CDIM;
    const __bf16* wvb = wv + (size_t)ntg * 64 * CDIM;
    v8f ck[4] = {};
    v8f cv[4] = {};
    for (int k0 = 0; k0 < CDIM; k0 += 32) {
        v16bf a = load_a_bf16(abase, CDIM, lane, k0);
        v16bf bk[4], bv[4];
#pragma unroll
        for (int u = 0; u < 4; ++u) {
            bk[u] = load_b_bf16(wkb + (size_t)u * 16 * CDIM, CDIM, lane, k0);
            bv[u] = load_b_bf16(wvb + (size_t)u * 16 * CDIM, CDIM, lane, k0);
        }
#pragma unroll
        for (int u = 0; u < 4; ++u) {
            ck[u] = wmma_bf16(a, bk[u], ck[u]);
            cv[u] = wmma_bf16(a, bv[u], cv[u]);
        }
    }
    int n = lane & 15, m0 = (lane >> 4) * 8;
#pragma unroll
    for (int u = 0; u < 4; ++u) {
        int nt = ntg * 4 + u;
        __bf16* kb = kout + ((size_t)b * SKV_P + jt * 16) * HID + nt * 16;
        __bf16* vb = vtout + ((size_t)b * HID + nt * 16 + n) * SKV_P + jt * 16;
#pragma unroll
        for (int i = 0; i < 8; ++i) {
            kb[(size_t)(m0 + i) * HID + n] = to_bf16(ck[u][i]);
            vb[m0 + i] = to_bf16(cv[u][i]);   // transposed store: contiguous
        }
    }
}

// ---------------------------------------------------------------------------
// K2: fully fused Q-proj + attention + O-proj. One block = (b, 16 q-rows),
// 4 waves, two head-passes (wave w -> heads w, w+4).
// Per pass:
//   Phase 1: q = hid_bf @ Wq_h^T * scale, 10 accs -> LDS bf16 [16x160]
//   Phase 2: scores = q @ k_h^T, 5 shared-A accs  -> LDS f32  [16x80]
//   Phase 3: masked softmax (j < 77)              -> LDS bf16 [16x96]
//   Phase 4: p @ v_h -> LDS attn tile [16x1280] (bf16) at cols h*160
// Finally (cooperative, all 4 waves):
//   Phase 5: out = atile @ Wo^T + bo -> global fp32
// Static LDS = 92KB (CDNA5 workgroup LDS allows up to 320KB).
// ---------------------------------------------------------------------------
__global__ __launch_bounds__(128)
void attn_fused_kernel(const __bf16* __restrict__ hiddenb, const __bf16* __restrict__ wq,
                       const __bf16* __restrict__ kws, const __bf16* __restrict__ vtws,
                       const __bf16* __restrict__ wo, const float* __restrict__ bo,
                       float* __restrict__ out) {
    __shared__ __bf16 qtile[4][16 * DHEAD];   // 20 KB
    __shared__ float  sc[4][16 * 80];         // 20 KB
    __shared__ __bf16 pr[4][16 * SKV_P];      // 12 KB
    __shared__ __bf16 atile[16 * HID];        // 40 KB

    int lane = threadIdx.x & 31;
    int wave = threadIdx.x >> 5;
    int st = blockIdx.x & 255;                   // [NB][SEQ/16]
    int b  = blockIdx.x >> 8;

    const float scale = 0.07905694150420949f;    // 1/sqrt(160)
    const __bf16* hbase = hiddenb + ((size_t)b * SEQ + (size_t)st * 16) * HID;
    int n = lane & 15, m0 = (lane >> 4) * 8;

    for (int pass = 0; pass < 2; ++pass) {
        int h = pass * 4 + wave;

        // ---- Phase 1: Q projection, 10 n-tile accumulators ----
        {
            v8f acc[10] = {};
            const __bf16* wb = wq + (size_t)h * DHEAD * HID;
            for (int k0 = 0; k0 < HID; k0 += 32) {
                v16bf a = load_a_bf16(hbase, HID, lane, k0);
#pragma unroll
                for (int g = 0; g < 2; ++g) {
                    v16bf bb[5];
#pragma unroll
                    for (int u = 0; u < 5; ++u)
                        bb[u] = load_b_bf16(wb + (size_t)(g * 5 + u) * 16 * HID, HID,
                                            lane, k0);
#pragma unroll
                    for (int u = 0; u < 5; ++u)
                        acc[g * 5 + u] = wmma_bf16(a, bb[u], acc[g * 5 + u]);
                }
            }
#pragma unroll
            for (int nt = 0; nt < 10; ++nt)
#pragma unroll
                for (int i = 0; i < 8; ++i)
                    qtile[wave][(m0 + i) * DHEAD + nt * 16 + n] =
                        to_bf16(acc[nt][i] * scale);
        }
        __syncthreads();

        // ---- Phase 2: scores = q @ k^T (K = d = 160), 5 jt-accumulators ----
        {
            v8f acc[5] = {};
            const __bf16* kb0 = kws + (size_t)b * SKV_P * HID + h * DHEAD;
#pragma unroll
            for (int k0 = 0; k0 < DHEAD; k0 += 32) {
                v16bf a = load_a_bf16(&qtile[wave][0], DHEAD, lane, k0);
                v16bf bb[5];
#pragma unroll
                for (int jt = 0; jt < 5; ++jt)
                    bb[jt] = load_b_bf16(kb0 + (size_t)jt * 16 * HID, HID, lane, k0);
#pragma unroll
                for (int jt = 0; jt < 5; ++jt)
                    acc[jt] = wmma_bf16(a, bb[jt], acc[jt]);
            }
#pragma unroll
            for (int jt = 0; jt < 5; ++jt)
#pragma unroll
                for (int i = 0; i < 8; ++i)
                    sc[wave][(m0 + i) * 80 + jt * 16 + n] = acc[jt][i];
        }
        __syncthreads();

        // ---- Phase 3: masked softmax over 77 valid kv tokens ----
        if (lane < 16) {
            float* row = &sc[wave][lane * 80];
            float mx = -3.402823466e38f;
            for (int j = 0; j < SKV; ++j) mx = fmaxf(mx, row[j]);
            float sum = 0.f;
            for (int j = 0; j < 80; ++j) {
                float e = (j < SKV) ? __expf(row[j] - mx) : 0.f;
                row[j] = e;
                sum += e;
            }
            float inv = 1.f / sum;
            __bf16* prow = &pr[wave][lane * SKV_P];
            for (int j = 0; j < 80; ++j) prow[j] = to_bf16(row[j] * inv);
            for (int j = 80; j < SKV_P; ++j) prow[j] = (__bf16)0.0f;
        }
        __syncthreads();

        // ---- Phase 4: p @ v -> LDS attn tile (cols h*160..h*160+159) ----
        {
            v8f acc[10] = {};
            const __bf16* vb0 = vtws + ((size_t)b * HID + h * DHEAD) * SKV_P;
#pragma unroll
            for (int k0 = 0; k0 < SKV_P; k0 += 32) {
                v16bf a = load_a_bf16(&pr[wave][0], SKV_P, lane, k0);
#pragma unroll
                for (int g = 0; g < 2; ++g) {
                    v16bf bb[5];
#pragma unroll
                    for (int u = 0; u < 5; ++u)
                        bb[u] = load_b_bf16(vb0 + (size_t)(g * 5 + u) * 16 * SKV_P,
                                            SKV_P, lane, k0);
#pragma unroll
                    for (int u = 0; u < 5; ++u)
                        acc[g * 5 + u] = wmma_bf16(a, bb[u], acc[g * 5 + u]);
                }
            }
#pragma unroll
            for (int nt = 0; nt < 10; ++nt)
#pragma unroll
                for (int i = 0; i < 8; ++i)
                    atile[(m0 + i) * HID + h * DHEAD + nt * 16 + n] =
                        to_bf16(acc[nt][i]);
        }
        __syncthreads();
    }

    // ---- Phase 5: cooperative O-projection out of the LDS attn tile ----
    // Wave w covers columns [w*320, (w+1)*320) as 4 groups of 5 n-tiles.
    for (int ng = 0; ng < 4; ++ng) {
        int nt0 = wave * 20 + ng * 5;
        const __bf16* wb = wo + (size_t)nt0 * 16 * HID;
        v8f acc[5] = {};
        for (int k0 = 0; k0 < HID; k0 += 32) {
            v16bf a = load_a_bf16(atile, HID, lane, k0);
            v16bf bb[5];
#pragma unroll
            for (int u = 0; u < 5; ++u)
                bb[u] = load_b_bf16(wb + (size_t)u * 16 * HID, HID, lane, k0);
#pragma unroll
            for (int u = 0; u < 5; ++u)
                acc[u] = wmma_bf16(a, bb[u], acc[u]);
        }
#pragma unroll
        for (int u = 0; u < 5; ++u) {
            int nt = nt0 + u;
            float bias = bo[nt * 16 + n];
            float* ob = out + ((size_t)b * SEQ + (size_t)st * 16) * HID + nt * 16;
#pragma unroll
            for (int i = 0; i < 8; ++i)
                ob[(size_t)(m0 + i) * HID + n] = acc[u][i] + bias;
        }
    }
}

// ---------------------------------------------------------------------------
extern "C" void kernel_launch(void* const* d_in, const int* in_sizes, int n_in,
                              void* d_out, int out_size, void* d_ws, size_t ws_size,
                              hipStream_t stream) {
    const float* hidden = (const float*)d_in[0];
    const float* enc    = (const float*)d_in[1];
    const float* Wq     = (const float*)d_in[2];
    const float* Wk     = (const float*)d_in[3];
    const float* Wv     = (const float*)d_in[4];
    const float* Wo     = (const float*)d_in[5];
    const float* bo     = (const float*)d_in[6];
    float* out = (float*)d_out;

    // workspace layout (bf16 everywhere, all offsets 256B aligned)
    char* ws = (char*)d_ws;
    size_t off = 0;
    __bf16* wq_bf  = (__bf16*)(ws + off); off += (size_t)HID * HID * 2;
    __bf16* wk_bf  = (__bf16*)(ws + off); off += (size_t)HID * CDIM * 2;
    __bf16* wv_bf  = (__bf16*)(ws + off); off += (size_t)HID * CDIM * 2;
    __bf16* wo_bf  = (__bf16*)(ws + off); off += (size_t)HID * HID * 2;
    __bf16* enc_bf = (__bf16*)(ws + off); off += (size_t)NB * SKV_P * CDIM * 2;
    __bf16* hid_bf = (__bf16*)(ws + off); off += (size_t)NB * SEQ * HID * 2;
    __bf16* k_bf   = (__bf16*)(ws + off); off += (size_t)NB * SKV_P * HID * 2;
    __bf16* vt_bf  = (__bf16*)(ws + off); off += (size_t)NB * HID * SKV_P * 2;

    // K0: one-time bf16 conversion (weights stay hot in 192MB L2 afterwards)
    cvt_bf16x4_kernel<<<256, 256, 0, stream>>>(Wq, wq_bf, (size_t)HID * HID / 4);
    cvt_bf16x4_kernel<<<256, 256, 0, stream>>>(Wk, wk_bf, (size_t)HID * CDIM / 4);
    cvt_bf16x4_kernel<<<256, 256, 0, stream>>>(Wv, wv_bf, (size_t)HID * CDIM / 4);
    cvt_bf16x4_kernel<<<256, 256, 0, stream>>>(Wo, wo_bf, (size_t)HID * HID / 4);
    cvt_enc_kernel<<<256, 256, 0, stream>>>(enc, enc_bf);
    cvt_bf16x4_kernel<<<4096, 256, 0, stream>>>(hidden, hid_bf,
                                                (size_t)NB * SEQ * HID / 4);

    // K1: K/V projections (k row-major, v transposed, zero-padded tokens)
    int k1_tasks = NB * 6 * 20;                          // 1920 waves
    kv_proj_kernel<<<k1_tasks / 4, 128, 0, stream>>>(enc_bf, wk_bf, wv_bf, k_bf, vt_bf);

    // K2: fully fused Q-proj + attention + O-proj
    int k2_blocks = NB * (SEQ / 16);                     // 4096 blocks
    attn_fused_kernel<<<k2_blocks, 128, 0, stream>>>(hid_bf, wq_bf, k_bf, vt_bf,
                                                     wo_bf, bo, out);
}
